// GraphSAGELayer_64192581206554
// MI455X (gfx1250) — matrix operand order repaired and checked by reading, not compiled, not run on hardware
//
#include <hip/hip_runtime.h>

typedef __attribute__((ext_vector_type(16))) _Float16 v16h;
typedef __attribute__((ext_vector_type(8)))  _Float16 v8h;
typedef __attribute__((ext_vector_type(4)))  _Float16 v4h;
typedef __attribute__((ext_vector_type(8)))  float    v8f;

#define N_NODES 50000
#define DIM     128          // IN_DIM == OUT_DIM
#define LDS_PITCH (DIM + 8)  // +8 f16 pad -> conflict-free 16-row reads

// ---------------------------------------------------------------------------
// Zero helper (int4 stores)
// ---------------------------------------------------------------------------
__global__ void sage_zero_i4(int4* __restrict__ p, int n4) {
  int i = blockIdx.x * blockDim.x + threadIdx.x;
  if (i < n4) p[i] = int4{0, 0, 0, 0};
}

// ---------------------------------------------------------------------------
// f32 -> f16 pack (float4 -> v4h per thread); used for h, W_self, W_neigh
// ---------------------------------------------------------------------------
__global__ void sage_pack_f16(const float* __restrict__ src,
                              _Float16* __restrict__ dst, int n4) {
  int i = blockIdx.x * blockDim.x + threadIdx.x;
  if (i < n4) {
    const float4 v = ((const float4*)src)[i];
    v4h o;
    o[0] = (_Float16)v.x; o[1] = (_Float16)v.y;
    o[2] = (_Float16)v.z; o[3] = (_Float16)v.w;
    ((v4h*)dst)[i] = o;
  }
}

// ---------------------------------------------------------------------------
// In-degree histogram (1 int atomic per edge)
// ---------------------------------------------------------------------------
__global__ void sage_hist(const int* __restrict__ dst, int* __restrict__ counts,
                          int n_edges) {
  int e = blockIdx.x * blockDim.x + threadIdx.x;
  if (e < n_edges) atomicAdd(&counts[dst[e]], 1);
}

// ---------------------------------------------------------------------------
// Exclusive scan of counts -> row_ptr[0..N_NODES]; single 1024-thread block
// ---------------------------------------------------------------------------
__global__ void sage_scan(const int* __restrict__ counts,
                          int* __restrict__ row_ptr, int n) {
  __shared__ int partials[1024];
  const int T = 1024;
  const int tid = threadIdx.x;
  const int chunk = (n + T - 1) / T;
  const int base = tid * chunk;

  int sum = 0;
  for (int i = 0; i < chunk; ++i) {
    int idx = base + i;
    if (idx < n) sum += counts[idx];
  }
  partials[tid] = sum;
  __syncthreads();
  for (int off = 1; off < T; off <<= 1) {
    int v = partials[tid];
    int add = (tid >= off) ? partials[tid - off] : 0;
    __syncthreads();
    partials[tid] = v + add;
    __syncthreads();
  }
  int run = (tid == 0) ? 0 : partials[tid - 1];
  for (int i = 0; i < chunk; ++i) {
    int idx = base + i;
    if (idx < n) { row_ptr[idx] = run; run += counts[idx]; }
  }
  if (tid == T - 1) row_ptr[n] = run;
}

// ---------------------------------------------------------------------------
// CSR fill: edge_src_sorted[row_ptr[d] + cursor[d]++] = src[e]
// ---------------------------------------------------------------------------
__global__ void sage_fill(const int* __restrict__ src, const int* __restrict__ dst,
                          const int* __restrict__ row_ptr, int* __restrict__ cursor,
                          int* __restrict__ edge_src_sorted, int n_edges) {
  int e = blockIdx.x * blockDim.x + threadIdx.x;
  if (e < n_edges) {
    int d = dst[e];
    int pos = row_ptr[d] + atomicAdd(&cursor[d], 1);
    edge_src_sorted[pos] = src[e];
  }
}

// ---------------------------------------------------------------------------
// Fused gather + WMMA GEMM.
//  Stage:  As = h16 tile (one b128/thread);  Am = per-node neighbor mean,
//          computed atomic-free from CSR (wave w owns rows 2w, 2w+1;
//          lane l owns 4 features; f32 accumulate; v_rcp for 1/deg).
//  Math :  out = relu(As @ Wself^T + Am @ Wneigh^T + bias) via 8 chained
//          v_wmma_f32_16x16x32_f16 per wave (one 16x16 output tile each).
//  Grid = 50000/16 = 3125 blocks exactly -> no predication -> EXEC all-1s.
// ---------------------------------------------------------------------------
__global__ void sage_fused(const _Float16* __restrict__ h16,
                           const int*      __restrict__ edge_src_sorted,
                           const int*      __restrict__ row_ptr,
                           const float*    __restrict__ deg,
                           const _Float16* __restrict__ w16s,
                           const _Float16* __restrict__ w16n,
                           const float*    __restrict__ bias,
                           float*          __restrict__ out) {
  __shared__ __align__(16) _Float16 As[16][LDS_PITCH];
  __shared__ __align__(16) _Float16 Am[16][LDS_PITCH];

  const int mtile = blockIdx.x;            // 0..3124
  const int wave  = threadIdx.x >> 5;      // 0..7 (= ntile)
  const int lane  = threadIdx.x & 31;
  const int r16   = lane & 15;
  const int khalf = lane >> 4;
  const int t     = threadIdx.x;

  // -- stage As: 16x128 f16 = 4KB, one b128 load+store per thread
  {
    const int row = t >> 4;                // (t*8)/128
    const int col = (t & 15) * 8;
    const v8h hv = *(const v8h*)(h16 + (size_t)(mtile * 16 + row) * DIM + col);
    *(v8h*)&As[row][col] = hv;
  }

  // -- gather: wave computes mean rows 2*wave, 2*wave+1 into Am
#pragma unroll
  for (int j = 0; j < 2; ++j) {
    const int row  = wave * 2 + j;
    const int node = mtile * 16 + row;
    const int beg = row_ptr[node];
    const int end = row_ptr[node + 1];     // wave-uniform -> scalar loop
    float4 acc = {0.f, 0.f, 0.f, 0.f};
    for (int i = beg; i < end; ++i) {
      const int s = edge_src_sorted[i];    // wave-uniform scalar load
      const v4h v = *(const v4h*)(h16 + (size_t)s * DIM + lane * 4);
      acc.x += (float)v[0]; acc.y += (float)v[1];
      acc.z += (float)v[2]; acc.w += (float)v[3];
    }
    const float inv = __builtin_amdgcn_rcpf(deg[node]);
    v4h m;
    m[0] = (_Float16)(acc.x * inv); m[1] = (_Float16)(acc.y * inv);
    m[2] = (_Float16)(acc.z * inv); m[3] = (_Float16)(acc.w * inv);
    *(v4h*)&Am[row][lane * 4] = m;
  }
  __syncthreads();

  // -- WMMA: A from LDS (contiguous K: 16 f16 per lane half per K-block),
  //          B direct from pre-packed f16 weights (16 contiguous f16)
  const int n = wave * 16 + r16;
  const _Float16* __restrict__ wsrow = w16s + (size_t)n * DIM;
  const _Float16* __restrict__ wnrow = w16n + (size_t)n * DIM;

  v8f acc = {};
#pragma unroll
  for (int kb = 0; kb < 4; ++kb) {
    const int ab = kb * 32 + khalf * 16;   // A: K = ab+e (e<16), contiguous
    const v8h alo = *(const v8h*)&As[r16][ab];
    const v8h ahi = *(const v8h*)&As[r16][ab + 8];
    const v8h mlo = *(const v8h*)&Am[r16][ab];
    const v8h mhi = *(const v8h*)&Am[r16][ab + 8];
    const v16h a_h = __builtin_shufflevector(alo, ahi,
        0,1,2,3,4,5,6,7,8,9,10,11,12,13,14,15);
    const v16h a_m = __builtin_shufflevector(mlo, mhi,
        0,1,2,3,4,5,6,7,8,9,10,11,12,13,14,15);
    const int bb = kb * 32 + khalf * 16;   // B: lanes 0-15 K 0-15, 16-31 K 16-31
    const v16h b_s = *(const v16h*)(wsrow + bb);
    const v16h b_n = *(const v16h*)(wnrow + bb);
    acc = __builtin_amdgcn_wmma_f32_16x16x32_f16(false, a_h, false, b_s,
                                                 (short)0, acc, false, false);
    acc = __builtin_amdgcn_wmma_f32_16x16x32_f16(false, a_m, false, b_n,
                                                 (short)0, acc, false, false);
  }

  // -- epilogue: D layout VGPR r -> row (r + 8*khalf), col = lane&15
  const int ocol = wave * 16 + r16;
  const float bv = bias[ocol];
#pragma unroll
  for (int r = 0; r < 8; ++r) {
    const int orow = mtile * 16 + r + 8 * khalf;
    out[(size_t)orow * DIM + ocol] = fmaxf(acc[r] + bv, 0.0f);
  }
}

// ---------------------------------------------------------------------------
// Inputs (setup_inputs order): h, src, dst, deg, W_self, W_neigh, bias
// ---------------------------------------------------------------------------
extern "C" void kernel_launch(void* const* d_in, const int* in_sizes, int n_in,
                              void* d_out, int out_size, void* d_ws, size_t ws_size,
                              hipStream_t stream) {
  const float* h       = (const float*)d_in[0];
  const int*   src     = (const int*)  d_in[1];
  const int*   dst     = (const int*)  d_in[2];
  const float* deg     = (const float*)d_in[3];
  const float* W_self  = (const float*)d_in[4];
  const float* W_neigh = (const float*)d_in[5];
  const float* bias    = (const float*)d_in[6];
  float*       out     = (float*)d_out;
  const int n_edges = in_sizes[1];

  // workspace layout (16B aligned)
  char* ws = (char*)d_ws;
  size_t off = 0;
  _Float16* h16  = (_Float16*)(ws + off); off += (size_t)N_NODES * DIM * 2;  // 12.8MB
  _Float16* w16s = (_Float16*)(ws + off); off += (size_t)DIM * DIM * 2;      // 32KB
  _Float16* w16n = (_Float16*)(ws + off); off += (size_t)DIM * DIM * 2;      // 32KB
  int* row_ptr = (int*)(ws + off); off += (size_t)(N_NODES + 1) * sizeof(int);
  off = (off + 15) & ~(size_t)15;
  int* counts  = (int*)(ws + off); off += (size_t)N_NODES * sizeof(int);
  int* cursor  = (int*)(ws + off); off += (size_t)N_NODES * sizeof(int);
  off = (off + 15) & ~(size_t)15;
  int* edge_srt = (int*)(ws + off); off += (size_t)n_edges * sizeof(int);

  // 1) pack h / W_self / W_neigh to f16
  {
    const int n4h = N_NODES * DIM / 4;                 // 1.6M
    sage_pack_f16<<<(n4h + 255) / 256, 256, 0, stream>>>(h, h16, n4h);
    const int n4w = DIM * DIM / 4;                     // 4096
    sage_pack_f16<<<(n4w + 255) / 256, 256, 0, stream>>>(W_self,  w16s, n4w);
    sage_pack_f16<<<(n4w + 255) / 256, 256, 0, stream>>>(W_neigh, w16n, n4w);
  }
  // 2) CSR build: zero counts+cursor -> histogram -> scan -> fill
  {
    const int n4 = (2 * N_NODES) / 4;
    sage_zero_i4<<<(n4 + 255) / 256, 256, 0, stream>>>((int4*)counts, n4);
  }
  sage_hist<<<(n_edges + 255) / 256, 256, 0, stream>>>(dst, counts, n_edges);
  sage_scan<<<1, 1024, 0, stream>>>(counts, row_ptr, N_NODES);
  sage_fill<<<(n_edges + 255) / 256, 256, 0, stream>>>(src, dst, row_ptr, cursor,
                                                       edge_srt, n_edges);
  // 3) fused gather + WMMA GEMM + bias + relu
  sage_fused<<<N_NODES / 16, 256, 0, stream>>>(h16, edge_srt, row_ptr, deg,
                                               w16s, w16n, bias, out);
}
